// Decoder_70128226009608
// MI455X (gfx1250) — matrix operand order repaired and checked by reading, compiled, not run on hardware
//
#include <hip/hip_runtime.h>
#include <hip/hip_bf16.h>
#include <stdint.h>

// ---------------------------------------------------------------------------
// CDNA5 (gfx1250) bf16 WMMA decoder:  embed -> Luong attention -> LSTM -> FC
// Shapes: V=32000 EMB=256 DU=512 B=32 TIN=64 TOUT=64
// ---------------------------------------------------------------------------

typedef __attribute__((ext_vector_type(16))) __bf16 v16bf;
typedef __attribute__((ext_vector_type(8)))  float  v8f;

#if defined(__has_builtin)
# if __has_builtin(__builtin_amdgcn_tensor_load_to_lds)
#  define HAS_TDM 1
# endif
#endif
#if __has_include(<hip/amd_detail/amd_gfx1250_TDM.h>)
# define TDM_ARGS6 1
#endif

#ifdef HAS_TDM
typedef __attribute__((ext_vector_type(4))) unsigned int u32x4;
typedef __attribute__((ext_vector_type(8))) int          i32x8;
typedef __attribute__((ext_vector_type(4))) int          i32x4;
#endif

__device__ __forceinline__ unsigned short f2bf(float f) {
    union { float f; unsigned u; } x; x.f = f;
    unsigned r = x.u + 0x7fffu + ((x.u >> 16) & 1u);   // round-to-nearest-even
    return (unsigned short)(r >> 16);
}

union FragU { v16bf v; uint4 u[2]; };

// A fragment (16x32 bf16, M x K): lane%16 = m, lane/16 selects k-half.
__device__ __forceinline__ v16bf load_a(const unsigned short* A, int lda, int row0, int k0) {
    int lane = threadIdx.x & 31;
    const unsigned short* p = A + (size_t)(row0 + (lane & 15)) * lda + k0 + ((lane >> 4) * 8);
    FragU r;
    r.u[0] = *(const uint4*)(p);
    r.u[1] = *(const uint4*)(p + 16);
    return r.v;
}

// B fragment (32x16 bf16, K x N): lane%16 = n, lane/16 selects k-half (same k order as A)
__device__ __forceinline__ v16bf load_b(const unsigned short* B, int ldb, int k0, int n0) {
    int lane = threadIdx.x & 31;
    const unsigned short* p = B + (size_t)(k0 + ((lane >> 4) * 8)) * ldb + n0 + (lane & 15);
    union { v16bf v; unsigned short s[16]; } r;
    for (int i = 0; i < 8; ++i) {
        r.s[i]     = p[(size_t)i * ldb];
        r.s[8 + i] = p[(size_t)(16 + i) * ldb];
    }
    return r.v;
}

__device__ __forceinline__ v8f wmma_bf16(v16bf a, v16bf b, v8f c) {
    return __builtin_amdgcn_wmma_f32_16x16x32_bf16(false, a, false, b, (short)0, c, false, false);
}

#ifdef HAS_TDM
// Issue one TDM 2D tile load: global (row-major, ldb elems/row) -> LDS
// (row-major tileRows x tileCols, contiguous). bf16 elements (data_size=2B).
__device__ __forceinline__ void tdm_stage_b(const unsigned short* gsrc, int ldb,
                                            unsigned short* lds_dst,
                                            unsigned tileRows, unsigned tileCols,
                                            unsigned tensorRows, unsigned tensorCols) {
    unsigned long long ga = (unsigned long long)(uintptr_t)gsrc;
    unsigned lds_off = (unsigned)(uintptr_t)lds_dst;   // generic addr low 32 = LDS offset
    u32x4 g0;
    g0[0] = 1u;                                        // count=1, user mode
    g0[1] = lds_off;                                   // lds_addr (bytes)
    g0[2] = (unsigned)(ga & 0xffffffffu);              // global_addr[31:0]
    g0[3] = (unsigned)((ga >> 32) & 0x01ffffffu) | 0x80000000u; // addr[56:32] | type=2
    i32x8 g1;
    g1[0] = (int)(1u << 16);                           // wg_mask=0, data_size=1 (2 bytes)
    g1[1] = (int)((tensorCols & 0xffffu) << 16);       // tensor_dim0[15:0] @ bits63:48
    g1[2] = (int)((tensorCols >> 16) | ((tensorRows & 0xffffu) << 16)); // dim0 hi | dim1 lo
    g1[3] = (int)((tensorRows >> 16) | ((tileCols & 0xffffu) << 16));   // dim1 hi | tile_dim0
    g1[4] = (int)(tileRows & 0xffffu);                 // tile_dim1 (tile_dim2=0)
    g1[5] = (int)(unsigned)ldb;                        // tensor_dim0_stride[31:0]
    g1[6] = 0;                                         // stride hi | dim1_stride lo
    g1[7] = 0;
    i32x4 z4 = {0, 0, 0, 0};
#ifdef TDM_ARGS6
    i32x8 z8 = {0, 0, 0, 0, 0, 0, 0, 0};
    __builtin_amdgcn_tensor_load_to_lds(g0, g1, z4, z4, z8, 0);
#else
    __builtin_amdgcn_tensor_load_to_lds(g0, g1, z4, z4, 0);
#endif
}
#endif

// ---------------------------------------------------------------------------
// Elementwise helpers
// ---------------------------------------------------------------------------
__global__ void k_cvt(const float* __restrict__ src, unsigned short* __restrict__ dst, int n) {
    int i = blockIdx.x * blockDim.x + threadIdx.x;
    if (i < n) dst[i] = f2bf(src[i]);
}

// enc [32][64][512] f32 -> encT [32][512][64] bf16  (B operand of scores GEMM)
__global__ void k_enc_t(const float* __restrict__ enc, unsigned short* __restrict__ encT) {
    int i = blockIdx.x * blockDim.x + threadIdx.x;
    if (i >= 32 * 64 * 512) return;
    int d = i & 511, s = (i >> 9) & 63, b = i >> 15;
    encT[((size_t)b * 512 + d) * 64 + s] = f2bf(enc[i]);
}

// x = emb[tokens] ; also place x into lstm_in columns [512,768)
__global__ void k_embed(const int* __restrict__ tok, const float* __restrict__ emb,
                        unsigned short* __restrict__ xbf, unsigned short* __restrict__ lstm_in) {
    int i = blockIdx.x * blockDim.x + threadIdx.x;   // 2048*256
    if (i >= 2048 * 256) return;
    int e = i & 255, r = i >> 8;                      // r = b*64 + t
    unsigned short v = f2bf(emb[(size_t)tok[r] * 256 + e]);
    xbf[i] = v;
    lstm_in[(size_t)r * 768 + 512 + e] = v;
}

__global__ void k_prep(const float* __restrict__ h0, const float* __restrict__ c0,
                       unsigned short* __restrict__ hbuf0, float* __restrict__ cbuf) {
    int i = blockIdx.x * blockDim.x + threadIdx.x;   // 32*512
    if (i >= 32 * 512) return;
    hbuf0[i] = f2bf(h0[i]);
    cbuf[i]  = c0[i];
}

// ---------------------------------------------------------------------------
// Small bf16 WMMA GEMM (no staging): each wave owns a 16x64 strip.
// ---------------------------------------------------------------------------
__global__ void k_gemm(const unsigned short* __restrict__ A, int lda,
                       const unsigned short* __restrict__ B, int ldb,
                       float* __restrict__ Cf, unsigned short* __restrict__ Cbf, int ldc,
                       const float* __restrict__ bias, int Mtiles, int wavesN, int K) {
    int wid = (blockIdx.x * blockDim.x + threadIdx.x) >> 5;
    if (wid >= Mtiles * wavesN) return;               // uniform per wave
    int mt = wid / wavesN, nc = wid % wavesN;
    int row0 = mt * 16, n0 = nc * 64;
    v8f acc[4] = {{}, {}, {}, {}};
    for (int k0 = 0; k0 < K; k0 += 32) {
        v16bf a = load_a(A, lda, row0, k0);
        acc[0] = wmma_bf16(a, load_b(B, ldb, k0, n0),      acc[0]);
        acc[1] = wmma_bf16(a, load_b(B, ldb, k0, n0 + 16), acc[1]);
        acc[2] = wmma_bf16(a, load_b(B, ldb, k0, n0 + 32), acc[2]);
        acc[3] = wmma_bf16(a, load_b(B, ldb, k0, n0 + 48), acc[3]);
    }
    int lane = threadIdx.x & 31;
    int col = lane & 15, mb = 8 * (lane >> 4);
    for (int j = 0; j < 4; ++j) {
        float bs = bias[n0 + j * 16 + col];
        for (int r = 0; r < 8; ++r) {
            float v = acc[j][r] + bs;
            size_t off = (size_t)(row0 + mb + r) * ldc + n0 + j * 16 + col;
            if (Cbf) Cbf[off] = f2bf(v);
            else     Cf[off]  = v;
        }
    }
}

// ---------------------------------------------------------------------------
// Large bf16 WMMA GEMM with LDS-staged B panels (8x reuse).
// Block: 256 threads = 8 waves; block tile = 128 rows x 64 cols.
// K consumed in 128-wide panels; B panel (128x64 bf16 = 16KB) staged to LDS,
// double-buffered, via TDM tensor_load_to_lds when available.
// ---------------------------------------------------------------------------
__global__ void k_gemm_lds(const unsigned short* __restrict__ A, int lda,
                           const unsigned short* __restrict__ B, int ldb,
                           float* __restrict__ Cf, unsigned short* __restrict__ Cbf, int ldc,
                           const float* __restrict__ bias, int wavesN, int K, int Nfull) {
    __shared__ unsigned short Bs[2][128][64];
    int rg = blockIdx.x / wavesN, nc = blockIdx.x % wavesN;
    int n0 = nc * 64;
    int w = threadIdx.x >> 5, lane = threadIdx.x & 31;
    int row0 = rg * 128 + w * 16;
    int panels = K >> 7;                 // K multiple of 128
    v8f acc[4] = {{}, {}, {}, {}};
#ifdef HAS_TDM
    if (w == 0)
        tdm_stage_b(B + n0, ldb, &Bs[0][0][0], 128, 64, (unsigned)K, (unsigned)Nfull);
#endif
    for (int p = 0; p < panels; ++p) {
        unsigned short (*buf)[64] = Bs[p & 1];
#ifdef HAS_TDM
        if (w == 0) {
            if (p + 1 < panels) {
                tdm_stage_b(B + (size_t)(p + 1) * 128 * ldb + n0, ldb,
                            &Bs[(p + 1) & 1][0][0], 128, 64, (unsigned)K, (unsigned)Nfull);
                __builtin_amdgcn_s_wait_tensorcnt(1);   // panel p landed
            } else {
                __builtin_amdgcn_s_wait_tensorcnt(0);
            }
        }
        __syncthreads();
#else
        for (int i = threadIdx.x; i < 1024; i += 256) { // 128 rows x 8 chunks of 16B
            int r = i >> 3, cq = (i & 7) * 8;
            *(uint4*)&buf[r][cq] =
                *(const uint4*)(B + (size_t)(p * 128 + r) * ldb + n0 + cq);
        }
        __syncthreads();
#endif
        const unsigned short* Bp = &buf[0][0];
        for (int kk = 0; kk < 128; kk += 32) {
            v16bf a = load_a(A, lda, row0, p * 128 + kk);
            acc[0] = wmma_bf16(a, load_b(Bp, 64, kk, 0),  acc[0]);
            acc[1] = wmma_bf16(a, load_b(Bp, 64, kk, 16), acc[1]);
            acc[2] = wmma_bf16(a, load_b(Bp, 64, kk, 32), acc[2]);
            acc[3] = wmma_bf16(a, load_b(Bp, 64, kk, 48), acc[3]);
        }
        __syncthreads();
    }
    int col = lane & 15, mb = 8 * (lane >> 4);
    for (int j = 0; j < 4; ++j) {
        float bs = bias[n0 + j * 16 + col];
        for (int r = 0; r < 8; ++r) {
            float v = acc[j][r] + bs;
            size_t off = (size_t)(row0 + mb + r) * ldc + n0 + j * 16 + col;
            if (Cbf) Cbf[off] = f2bf(v);
            else     Cf[off]  = v;
        }
    }
}

// ---------------------------------------------------------------------------
// Attention: one block (4 waves) per batch.
// ---------------------------------------------------------------------------
__global__ void k_attn(const unsigned short* __restrict__ q_bf,
                       const unsigned short* __restrict__ enc_bf,
                       const unsigned short* __restrict__ encT_bf,
                       unsigned short* __restrict__ lstm_in,
                       float* __restrict__ attn_out) {
    __shared__ float          sc[64][64];
    __shared__ unsigned short aw[64][64];
    int b = blockIdx.x;
    int w = threadIdx.x >> 5, lane = threadIdx.x & 31;
    const unsigned short* Q  = q_bf    + (size_t)b * 64 * 512;
    const unsigned short* E  = enc_bf  + (size_t)b * 64 * 512;
    const unsigned short* ET = encT_bf + (size_t)b * 512 * 64;

    { // scores: wave w owns t-rows [16w,16w+16)
        v8f acc[4] = {{}, {}, {}, {}};
        int row0 = w * 16;
        for (int k0 = 0; k0 < 512; k0 += 32) {
            v16bf a = load_a(Q, 512, row0, k0);
            for (int nt = 0; nt < 4; ++nt)
                acc[nt] = wmma_bf16(a, load_b(ET, 64, k0, nt * 16), acc[nt]);
        }
        int col = lane & 15, mb = 8 * (lane >> 4);
        for (int nt = 0; nt < 4; ++nt)
            for (int r = 0; r < 8; ++r)
                sc[row0 + mb + r][nt * 16 + col] = acc[nt][r];
    }
    __syncthreads();
    if (threadIdx.x < 64) { // softmax over s for row t
        int t = threadIdx.x;
        float mx = -1e30f;
        for (int s = 0; s < 64; ++s) mx = fmaxf(mx, sc[t][s]);
        float sum = 0.f;
        for (int s = 0; s < 64; ++s) { float e = __expf(sc[t][s] - mx); sc[t][s] = e; sum += e; }
        float inv = 1.0f / sum;
        for (int s = 0; s < 64; ++s) {
            float wv = sc[t][s] * inv;
            attn_out[((size_t)b * 64 + t) * 64 + s] = wv;
            aw[t][s] = f2bf(wv);
        }
    }
    __syncthreads();
    { // att = aw (64x64) @ enc (64x512): wave w owns n-range [128w,128w+128)
        int nbase = w * 128;
        for (int mt = 0; mt < 4; ++mt) {
            for (int nt = 0; nt < 8; ++nt) {
                v8f acc = {};
                int n0 = nbase + nt * 16;
                for (int k0 = 0; k0 < 64; k0 += 32)
                    acc = wmma_bf16(load_a(&aw[0][0], 64, mt * 16, k0),
                                    load_b(E, 512, k0, n0), acc);
                int col = lane & 15, mb = 8 * (lane >> 4);
                for (int r = 0; r < 8; ++r) {
                    int t = mt * 16 + mb + r;
                    lstm_in[((size_t)b * 64 + t) * 768 + n0 + col] = f2bf(acc[r]);
                }
            }
        }
    }
}

// ---------------------------------------------------------------------------
// One LSTM timestep: z = Zx[:,t,:] + h @ Wh ; gate math ; h,c update.
// ---------------------------------------------------------------------------
__global__ void k_lstm_step(const unsigned short* __restrict__ hA,
                            unsigned short* __restrict__ hB,
                            const unsigned short* __restrict__ Wh_bf,
                            const float* __restrict__ Zx,
                            float* __restrict__ cbuf,
                            unsigned short* __restrict__ hs_bf,
                            float* __restrict__ out_h, float* __restrict__ out_c, int t) {
    __shared__ float zs[4][32][32];                  // [gate][batch][du_local]
    int blk = blockIdx.x;                            // du chunk of 32
    int w = threadIdx.x >> 5, lane = threadIdx.x & 31;
    int g = w >> 1, nt = w & 1;
    int n0 = g * 512 + blk * 32 + nt * 16;
    v8f acc0 = {}, acc1 = {};
    for (int k0 = 0; k0 < 512; k0 += 32) {
        v16bf bf = load_b(Wh_bf, 2048, k0, n0);
        acc0 = wmma_bf16(load_a(hA, 512, 0,  k0), bf, acc0);
        acc1 = wmma_bf16(load_a(hA, 512, 16, k0), bf, acc1);
    }
    int col = lane & 15, mb = 8 * (lane >> 4);
    for (int r = 0; r < 8; ++r) {
        zs[g][mb + r][nt * 16 + col]      = acc0[r];
        zs[g][16 + mb + r][nt * 16 + col] = acc1[r];
    }
    __syncthreads();
    for (int i = threadIdx.x; i < 1024; i += blockDim.x) {
        int bb = i >> 5, dl = i & 31;
        int du = blk * 32 + dl;
        size_t zr = ((size_t)bb * 64 + t) * 2048;
        float zi = zs[0][bb][dl] + Zx[zr + du];
        float zf = zs[1][bb][dl] + Zx[zr + 512 + du];
        float zg = zs[2][bb][dl] + Zx[zr + 1024 + du];
        float zo = zs[3][bb][dl] + Zx[zr + 1536 + du];
        float si = 1.f / (1.f + __expf(-zi));
        float sf = 1.f / (1.f + __expf(-zf));
        float so = 1.f / (1.f + __expf(-zo));
        float c = sf * cbuf[bb * 512 + du] + si * tanhf(zg);
        float h = so * tanhf(c);
        cbuf[bb * 512 + du] = c;
        hB[bb * 512 + du] = f2bf(h);
        hs_bf[((size_t)bb * 64 + t) * 512 + du] = f2bf(h);
        if (t == 63) { out_h[bb * 512 + du] = h; out_c[bb * 512 + du] = c; }
    }
}

// ---------------------------------------------------------------------------
extern "C" void kernel_launch(void* const* d_in, const int* in_sizes, int n_in,
                              void* d_out, int out_size, void* d_ws, size_t ws_size,
                              hipStream_t stream) {
    const int*   tok = (const int*)  d_in[0];
    const float* enc = (const float*)d_in[1];
    const float* h0  = (const float*)d_in[2];
    const float* c0  = (const float*)d_in[3];
    const float* emb = (const float*)d_in[4];
    const float* Wq  = (const float*)d_in[5];
    const float* bq  = (const float*)d_in[6];
    const float* Wx  = (const float*)d_in[7];
    const float* Wh  = (const float*)d_in[8];
    const float* bl  = (const float*)d_in[9];
    const float* Wfc = (const float*)d_in[10];
    const float* bfc = (const float*)d_in[11];
    float* out = (float*)d_out;
    float* out_logits = out;                              // 2048*32000
    float* out_h      = out + (size_t)2048 * 32000;       // 32*512
    float* out_c      = out_h + 32 * 512;                 // 32*512
    float* out_attn   = out_c + 32 * 512;                 // 32*64*64

    char* ws = (char*)d_ws;
    size_t off = 0;
    auto alloc = [&](size_t bytes) { char* p = ws + off; off += (bytes + 255) & ~(size_t)255; return p; };
    unsigned short* enc_bf  = (unsigned short*)alloc((size_t)32 * 64 * 512 * 2);
    unsigned short* encT_bf = (unsigned short*)alloc((size_t)32 * 512 * 64 * 2);
    unsigned short* xbf     = (unsigned short*)alloc((size_t)2048 * 256 * 2);
    unsigned short* Wq_bf   = (unsigned short*)alloc((size_t)256 * 512 * 2);
    unsigned short* Wx_bf   = (unsigned short*)alloc((size_t)768 * 2048 * 2);
    unsigned short* Wh_bf   = (unsigned short*)alloc((size_t)512 * 2048 * 2);
    unsigned short* Wfc_bf  = (unsigned short*)alloc((size_t)512 * 32000 * 2);
    unsigned short* q_bf    = (unsigned short*)alloc((size_t)2048 * 512 * 2);
    unsigned short* lstm_in = (unsigned short*)alloc((size_t)2048 * 768 * 2);
    float*          Zx      = (float*)        alloc((size_t)2048 * 2048 * 4);
    unsigned short* hs_bf   = (unsigned short*)alloc((size_t)2048 * 512 * 2);
    unsigned short* hbuf    = (unsigned short*)alloc((size_t)2 * 32 * 512 * 2);
    float*          cbuf    = (float*)        alloc((size_t)32 * 512 * 4);

    auto cvt = [&](const float* s, unsigned short* d, int n) {
        k_cvt<<<(n + 255) / 256, 256, 0, stream>>>(s, d, n);
    };
    // 1) weight / activation conversions to bf16
    cvt(Wq,  Wq_bf,  256 * 512);
    cvt(Wx,  Wx_bf,  768 * 2048);
    cvt(Wh,  Wh_bf,  512 * 2048);
    cvt(Wfc, Wfc_bf, 512 * 32000);
    cvt(enc, enc_bf, 32 * 64 * 512);
    k_enc_t<<<(32 * 64 * 512 + 255) / 256, 256, 0, stream>>>(enc, encT_bf);

    // 2) embedding gather (also fills lstm_in[:,512:768])
    k_embed<<<(2048 * 256 + 255) / 256, 256, 0, stream>>>(tok, emb, xbf, lstm_in);

    // 3) q = x @ Wq + bq     [2048,256]x[256,512] -> bf16
    k_gemm<<<(128 * 8 * 32) / 256, 256, 0, stream>>>(xbf, 256, Wq_bf, 512,
                                                     nullptr, q_bf, 512, bq, 128, 8, 256);

    // 4) attention per batch (fills lstm_in[:,0:512] and attn_w output)
    k_attn<<<32, 128, 0, stream>>>(q_bf, enc_bf, encT_bf, lstm_in, out_attn);

    // 5) Zx = lstm_in @ Wx + b_lstm   [2048,768]x[768,2048] -> f32 (LDS/TDM staged)
    k_gemm_lds<<<16 * 32, 256, 0, stream>>>(lstm_in, 768, Wx_bf, 2048,
                                            Zx, nullptr, 2048, bl, 32, 768, 2048);

    // 6) initial state
    k_prep<<<(32 * 512 + 255) / 256, 256, 0, stream>>>(h0, c0, hbuf, cbuf);

    // 7) sequential LSTM: 64 graph-captured launches; h double-buffered
    for (int t = 0; t < 64; ++t) {
        unsigned short* hA = hbuf + (size_t)(t & 1) * 32 * 512;
        unsigned short* hB = hbuf + (size_t)((t + 1) & 1) * 32 * 512;
        k_lstm_step<<<16, 256, 0, stream>>>(hA, hB, Wh_bf, Zx, cbuf, hs_bf, out_h, out_c, t);
    }

    // 8) logits = hs @ Wfc + bfc   [2048,512]x[512,32000] -> f32 (LDS/TDM staged)
    k_gemm_lds<<<16 * 500, 256, 0, stream>>>(hs_bf, 512, Wfc_bf, 32000,
                                             out_logits, nullptr, 32000, bfc, 500, 512, 32000);
    (void)in_sizes; (void)n_in; (void)out_size; (void)ws_size;
}
